// AttnRCR_89361089561152
// MI455X (gfx1250) — matrix operand order repaired and checked by reading, not compiled, not run on hardware
//
#include <hip/hip_runtime.h>
#include <hip/hip_bf16.h>

// ---------------- problem constants ----------------
constexpr int BB  = 128;   // batch
constexpr int NQ  = 256;
constexpr int NC  = 128;
constexpr int DD  = 1024;
constexpr int AD  = 256;   // ATTN_DIM
constexpr int AD2 = 512;   // 2*ATTN_DIM
constexpr int ADH = 128;   // ATTN_DIM/2
constexpr int NL  = 2;

constexpr int ROWS = BB * NQ;          // 32768 "token" rows
typedef __bf16 bf16;

typedef __attribute__((ext_vector_type(16))) __bf16 v16bf;
typedef __attribute__((ext_vector_type(8)))  __bf16 v8bf;
typedef __attribute__((ext_vector_type(8)))  float  v8f;

// ---------------- workspace layout (bytes) ----------------
constexpr size_t SZ_CTXB = (size_t)BB*NC*DD*2;      // context bf16 (c-major)
constexpr size_t SZ_CTXT = (size_t)BB*DD*NC*2;      // context^T bf16 (d-major)
constexpr size_t SZ_QM   = (size_t)BB*NQ*DD*2;      // bf16(query*matrix)
constexpr size_t SZ_MAT  = (size_t)BB*NQ*DD*4;      // matrix f32
constexpr size_t SZ_T    = (size_t)BB*NQ*NC*4;      // leaky scores^T f32
constexpr size_t SZ_RN   = (size_t)BB*NC*4;         // 1/(||.||+eps) per (b,c)
constexpr size_t SZ_SM   = (size_t)BB*NQ*4;         // smooth f32
constexpr size_t SZ_ATT  = (size_t)BB*NQ*NC*2;      // attn bf16
constexpr size_t SZ_D2   = (size_t)BB*NQ*DD*2;      // (q-wc)^2 bf16
constexpr size_t SZ_CP   = (size_t)ROWS*AD*4;       // common pre-norm f32
constexpr size_t SZ_CB   = (size_t)ROWS*AD*2;       // common bf16
constexpr size_t SZ_H1   = (size_t)ROWS*AD2*2;      // tanh hidden bf16
constexpr size_t SZ_HS   = (size_t)ROWS*ADH*2;      // smooth hidden bf16
constexpr size_t SZ_CWT  = (size_t)NL*AD*DD*2;      // cw_W^T bf16
constexpr size_t SZ_MW1T = (size_t)NL*AD2*AD*2;     // mw_W1^T
constexpr size_t SZ_MW2T = (size_t)NL*DD*AD2*2;     // mw_W2^T
constexpr size_t SZ_SW1T = (size_t)NL*ADH*AD*2;     // sw_W1^T

constexpr size_t OFF_CTXB = 0;
constexpr size_t OFF_CTXT = OFF_CTXB + SZ_CTXB;
constexpr size_t OFF_QM   = OFF_CTXT + SZ_CTXT;
constexpr size_t OFF_MAT  = OFF_QM   + SZ_QM;
constexpr size_t OFF_T    = OFF_MAT  + SZ_MAT;
constexpr size_t OFF_RN   = OFF_T    + SZ_T;
constexpr size_t OFF_SM   = OFF_RN   + SZ_RN;
constexpr size_t OFF_ATT  = OFF_SM   + SZ_SM;
constexpr size_t OFF_D2   = OFF_ATT  + SZ_ATT;
constexpr size_t OFF_CP   = OFF_D2   + SZ_D2;
constexpr size_t OFF_CB   = OFF_CP   + SZ_CP;
constexpr size_t OFF_H1   = OFF_CB   + SZ_CB;
constexpr size_t OFF_HS   = OFF_H1   + SZ_H1;
constexpr size_t OFF_CWT  = OFF_HS   + SZ_HS;
constexpr size_t OFF_MW1T = OFF_CWT  + SZ_CWT;
constexpr size_t OFF_MW2T = OFF_MW1T + SZ_MW1T;
constexpr size_t OFF_SW1T = OFF_MW2T + SZ_MW2T;

// ---------------- wave helpers ----------------
__device__ __forceinline__ float wave_sum(float v) {
  #pragma unroll
  for (int m = 16; m >= 1; m >>= 1) v += __shfl_xor(v, m, 32);
  return v;
}
__device__ __forceinline__ float wave_max(float v) {
  #pragma unroll
  for (int m = 16; m >= 1; m >>= 1) v = fmaxf(v, __shfl_xor(v, m, 32));
  return v;
}

// ---------------- WMMA GEMM core (software pipelined) ----------------
// Wave computes a 16-row x (NT*16)-col strip of C = A(bf16, MxK row-major) *
// BT^T (BT is bf16, N-major / K-contiguous), accumulating in f32.
// A fragment (16x32): lane(m=lane&15, h=lane>>4) holds
//   K = {h*16 .. h*16+15}  -> two contiguous v8bf loads.
// B fragment (32x16): lane(n=lane&15, h=lane>>4) holds K = {h*16 .. h*16+15}
//   -> one contiguous v16bf load from BT row n.
// Pipeline: A double-buffered across k-steps; B uses a distance-2 rolling
// buffer so each WMMA consumes a fragment whose load was issued two WMMAs
// earlier (loads overlap matrix-pipe execution instead of wait-0 stalls).
template<int NT, int K>
__device__ __forceinline__ void gemm_strip(const bf16* __restrict__ A, int lda,
                                           const bf16* __restrict__ BT, int ldb,
                                           v8f acc[NT]) {
  static_assert(NT >= 4, "pipeline assumes NT >= 4");
  #pragma unroll
  for (int t = 0; t < NT; ++t)
    #pragma unroll
    for (int j = 0; j < 8; ++j) acc[t][j] = 0.0f;

  const int lane = threadIdx.x & 31;
  const int m    = lane & 15;
  const int h    = lane >> 4;
  const bf16* ap = A  + (size_t)m * lda + h * 16;
  const bf16* bp = BT + (size_t)m * ldb + h * 16;   // m==n for B lanes
  const size_t bstr = (size_t)16 * ldb;             // stride between col-tiles

  auto loadA = [&](int k0) {
    v16bf a;
    *((v8bf*)&a)     = *(const v8bf*)(ap + k0);
    *((v8bf*)&a + 1) = *(const v8bf*)(ap + k0 + 8);
    return a;
  };
  auto loadB = [&](int t, int k0) {
    return *(const v16bf*)(bp + (size_t)t * bstr + k0);
  };

  constexpr int STEPS = K / 32;
  v16bf a_cur = loadA(0);
  v16bf b0 = loadB(0, 0);
  v16bf b1 = loadB(1, 0);

  #pragma unroll 1
  for (int s = 0; s < STEPS; ++s) {
    const int k0 = s * 32;
    const int kn = (s + 1 < STEPS) ? k0 + 32 : 0;   // last step: benign reload
    v16bf a_nxt = loadA(kn);
    __builtin_prefetch(ap + k0 + 128, 0, 3);
    #pragma unroll
    for (int t = 0; t < NT; ++t) {
      v16bf b_nxt;
      if (t + 2 < NT)        b_nxt = loadB(t + 2, k0);
      else if (t + 2 == NT)  b_nxt = loadB(0, kn);
      else                   b_nxt = loadB(1, kn);
      acc[t] = __builtin_amdgcn_wmma_f32_16x16x32_bf16(
          false, a_cur, false, b0, (short)0, acc[t], false, false);
      b0 = b1;
      b1 = b_nxt;
    }
    a_cur = a_nxt;
  }
}

// ---------------- elementwise / prep kernels ----------------
__global__ void k_init(float* __restrict__ matrix, float* __restrict__ smooth) {
  size_t tid = (size_t)blockIdx.x * blockDim.x + threadIdx.x;
  size_t stride = (size_t)gridDim.x * blockDim.x;
  for (size_t i = tid; i < (size_t)BB*NQ*DD; i += stride) matrix[i] = 1.0f;
  for (size_t i = tid; i < (size_t)BB*NQ;     i += stride) smooth[i] = 10.0f;
}

__global__ void k_cvt_ctx(const float* __restrict__ ctx,
                          bf16* __restrict__ ctxb, bf16* __restrict__ ctxT) {
  size_t i = (size_t)blockIdx.x * blockDim.x + threadIdx.x;
  if (i >= (size_t)BB*NC*DD) return;
  int b = (int)(i / (NC*DD));
  int r = (int)(i % (NC*DD));
  int c = r / DD, d = r % DD;
  float v = ctx[i];
  ctxb[i] = (bf16)v;
  ctxT[(size_t)b*DD*NC + (size_t)d*NC + c] = (bf16)v;
}

__global__ void k_cvt_w(const float* __restrict__ cw,  const float* __restrict__ mw1,
                        const float* __restrict__ mw2, const float* __restrict__ sw1,
                        bf16* __restrict__ cwT, bf16* __restrict__ mw1T,
                        bf16* __restrict__ mw2T, bf16* __restrict__ sw1T) {
  const size_t S1 = (size_t)NL*DD*AD, S2 = (size_t)NL*AD*AD2,
               S3 = (size_t)NL*AD2*DD, S4 = (size_t)NL*AD*ADH;
  size_t i = (size_t)blockIdx.x * blockDim.x + threadIdx.x;
  if (i < S1) {
    size_t l = i / (DD*AD), r = i % (DD*AD), k = r / AD, n = r % AD;
    cwT[(l*AD + n)*DD + k] = (bf16)cw[i];
  } else if (i < S1 + S2) {
    size_t j = i - S1, l = j / (AD*AD2), r = j % (AD*AD2), k = r / AD2, n = r % AD2;
    mw1T[(l*AD2 + n)*AD + k] = (bf16)mw1[j];
  } else if (i < S1 + S2 + S3) {
    size_t j = i - S1 - S2, l = j / (AD2*DD), r = j % (AD2*DD), k = r / DD, n = r % DD;
    mw2T[(l*DD + n)*AD2 + k] = (bf16)mw2[j];
  } else if (i < S1 + S2 + S3 + S4) {
    size_t j = i - S1 - S2 - S3, l = j / (AD*ADH), r = j % (AD*ADH), k = r / ADH, n = r % ADH;
    sw1T[(l*ADH + n)*AD + k] = (bf16)sw1[j];
  }
}

__global__ void k_qm(const float* __restrict__ q, const float* __restrict__ matrix,
                     bf16* __restrict__ qm) {
  size_t i = (size_t)blockIdx.x * blockDim.x + threadIdx.x;
  if (i < (size_t)BB*NQ*DD) qm[i] = (bf16)(q[i] * matrix[i]);
}

// ---------------- attention scan ----------------
// scores^T with fused LeakyReLU(0.1): T[b,q,c] = leaky(ctx[b,c,:].qm[b,q,:])
__global__ void __launch_bounds__(256)
k_scores(const bf16* __restrict__ qm, const bf16* __restrict__ ctxb,
         float* __restrict__ T) {
  int w  = blockIdx.x * (blockDim.x >> 5) + (threadIdx.x >> 5);  // 2048 waves
  int b  = w >> 4, qt = w & 15;
  const bf16* A  = qm   + ((size_t)b*NQ + qt*16) * DD;
  const bf16* BT = ctxb + (size_t)b*NC*DD;
  v8f acc[8];
  gemm_strip<8, DD>(A, DD, BT, DD, acc);
  int lane = threadIdx.x & 31, n = lane & 15, g = lane >> 4;
  float* Cp = T + ((size_t)b*NQ + qt*16) * NC + n;
  #pragma unroll
  for (int t = 0; t < 8; ++t)
    #pragma unroll
    for (int i = 0; i < 8; ++i) {
      float v = acc[t][i];
      Cp[(size_t)(g*8 + i)*NC + t*16] = (v >= 0.0f) ? v : 0.1f*v;
    }
}

// 1/(sqrt(sum_q T^2)+eps) per (b,c)  -- l2norm over nq axis
__global__ void k_rnorm(const float* __restrict__ T, float* __restrict__ rn) {
  int b = blockIdx.x, c = threadIdx.x;
  float s = 0.0f;
  for (int q = 0; q < NQ; ++q) {
    float x = T[((size_t)b*NQ + q)*NC + c];
    s += x*x;
  }
  rn[b*NC + c] = 1.0f / (sqrtf(s) + 1e-8f);
}

// softmax over nc with per-row smooth; one wave per (b,q) row
__global__ void __launch_bounds__(256)
k_softmax(const float* __restrict__ T, const float* __restrict__ rn,
          const float* __restrict__ smooth, bf16* __restrict__ attnb,
          float* __restrict__ outf) {
  int row  = blockIdx.x * (blockDim.x >> 5) + (threadIdx.x >> 5);
  if (row >= ROWS) return;
  int lane = threadIdx.x & 31;
  int b = row / NQ;
  const float* t = T + (size_t)row * NC;
  float sm = smooth[row];
  float v[4], mx = -3.0e38f;
  #pragma unroll
  for (int i = 0; i < 4; ++i) {
    int c = lane + i*32;
    v[i] = t[c] * rn[b*NC + c] * sm;
    mx = fmaxf(mx, v[i]);
  }
  mx = wave_max(mx);
  float s = 0.0f;
  #pragma unroll
  for (int i = 0; i < 4; ++i) { v[i] = __expf(v[i] - mx); s += v[i]; }
  s = wave_sum(s);
  float inv = 1.0f / s;
  #pragma unroll
  for (int i = 0; i < 4; ++i) {
    int c = lane + i*32;
    float a = v[i] * inv;
    attnb[(size_t)row*NC + c] = (bf16)a;
    if (outf) outf[(size_t)row*NC + c] = a;
  }
}

// wc = attn @ ctx ; epilogue: mode 0 -> diff2 = bf16((q - wc)^2), mode 1 -> out_wc
__global__ void __launch_bounds__(256)
k_wc(const bf16* __restrict__ attnb, const bf16* __restrict__ ctxT,
     const float* __restrict__ query, bf16* __restrict__ d2,
     float* __restrict__ out_wc, int mode) {
  int w  = blockIdx.x * (blockDim.x >> 5) + (threadIdx.x >> 5);  // 16384 waves
  int b  = w >> 7, r = w & 127, qt = r >> 3, cs = r & 7;
  const bf16* A  = attnb + ((size_t)b*NQ + qt*16) * NC;
  const bf16* BT = ctxT  + (size_t)b*DD*NC + (size_t)cs*128*NC;
  v8f acc[8];
  gemm_strip<8, NC>(A, NC, BT, NC, acc);
  int lane = threadIdx.x & 31, n = lane & 15, g = lane >> 4;
  #pragma unroll
  for (int t = 0; t < 8; ++t)
    #pragma unroll
    for (int i = 0; i < 8; ++i) {
      int rowg = b*NQ + qt*16 + g*8 + i;
      int col  = cs*128 + t*16 + n;
      size_t idx = (size_t)rowg * DD + col;
      float wcv = acc[t][i];
      if (mode == 0) { float d = query[idx] - wcv; d2[idx] = (bf16)(d*d); }
      else           { out_wc[idx] = wcv; }
    }
}

// ---------------- similarity / MLP stages ----------------
__global__ void __launch_bounds__(256)
k_common(const bf16* __restrict__ d2, const bf16* __restrict__ cwT,
         const float* __restrict__ bias, float* __restrict__ cp) {
  int w  = blockIdx.x * (blockDim.x >> 5) + (threadIdx.x >> 5);  // 4096 waves
  int rt = w >> 1, cs = w & 1;
  const bf16* A  = d2  + (size_t)rt*16*DD;
  const bf16* BT = cwT + (size_t)cs*128*DD;
  v8f acc[8];
  gemm_strip<8, DD>(A, DD, BT, DD, acc);
  int lane = threadIdx.x & 31, n = lane & 15, g = lane >> 4;
  #pragma unroll
  for (int t = 0; t < 8; ++t)
    #pragma unroll
    for (int i = 0; i < 8; ++i) {
      int col = cs*128 + t*16 + n;
      cp[(size_t)(rt*16 + g*8 + i)*AD + col] = acc[t][i] + bias[col];
    }
}

__global__ void __launch_bounds__(256)
k_l2common(const float* __restrict__ cp, bf16* __restrict__ cb) {
  int row  = blockIdx.x * (blockDim.x >> 5) + (threadIdx.x >> 5);
  if (row >= ROWS) return;
  int lane = threadIdx.x & 31;
  const float* p = cp + (size_t)row * AD;
  float v[8], s = 0.0f;
  #pragma unroll
  for (int i = 0; i < 8; ++i) { v[i] = p[lane + 32*i]; s += v[i]*v[i]; }
  s = wave_sum(s);
  float inv = 1.0f / (sqrtf(s) + 1e-8f);
  #pragma unroll
  for (int i = 0; i < 8; ++i) cb[(size_t)row*AD + lane + 32*i] = (bf16)(v[i]*inv);
}

__global__ void __launch_bounds__(256)
k_h1(const bf16* __restrict__ cb, const bf16* __restrict__ mw1T,
     const float* __restrict__ b1, bf16* __restrict__ h1) {
  int w  = blockIdx.x * (blockDim.x >> 5) + (threadIdx.x >> 5);  // 8192 waves
  int rt = w >> 2, cs = w & 3;
  const bf16* A  = cb   + (size_t)rt*16*AD;
  const bf16* BT = mw1T + (size_t)cs*128*AD;
  v8f acc[8];
  gemm_strip<8, AD>(A, AD, BT, AD, acc);
  int lane = threadIdx.x & 31, n = lane & 15, g = lane >> 4;
  #pragma unroll
  for (int t = 0; t < 8; ++t)
    #pragma unroll
    for (int i = 0; i < 8; ++i) {
      int col = cs*128 + t*16 + n;
      h1[(size_t)(rt*16 + g*8 + i)*AD2 + col] = (bf16)tanhf(acc[t][i] + b1[col]);
    }
}

__global__ void __launch_bounds__(256)
k_matrix(const bf16* __restrict__ h1, const bf16* __restrict__ mw2T,
         const float* __restrict__ b2, float* __restrict__ matrix) {
  int w  = blockIdx.x * (blockDim.x >> 5) + (threadIdx.x >> 5);  // 16384 waves
  int rt = w >> 3, cs = w & 7;
  const bf16* A  = h1   + (size_t)rt*16*AD2;
  const bf16* BT = mw2T + (size_t)cs*128*AD2;
  v8f acc[8];
  gemm_strip<8, AD2>(A, AD2, BT, AD2, acc);
  int lane = threadIdx.x & 31, n = lane & 15, g = lane >> 4;
  #pragma unroll
  for (int t = 0; t < 8; ++t)
    #pragma unroll
    for (int i = 0; i < 8; ++i) {
      int col = cs*128 + t*16 + n;
      size_t idx = (size_t)(rt*16 + g*8 + i)*DD + col;
      float m = tanhf(acc[t][i] + b2[col]) + matrix[idx];
      matrix[idx] = fminf(1.0f, fmaxf(-1.0f, m));
    }
}

__global__ void __launch_bounds__(256)
k_hs(const bf16* __restrict__ cb, const bf16* __restrict__ sw1T,
     const float* __restrict__ b1, bf16* __restrict__ hs) {
  int w  = blockIdx.x * (blockDim.x >> 5) + (threadIdx.x >> 5);  // 2048 waves
  int rt = w;
  const bf16* A  = cb + (size_t)rt*16*AD;
  v8f acc[8];
  gemm_strip<8, AD>(A, AD, sw1T, AD, acc);
  int lane = threadIdx.x & 31, n = lane & 15, g = lane >> 4;
  #pragma unroll
  for (int t = 0; t < 8; ++t)
    #pragma unroll
    for (int i = 0; i < 8; ++i) {
      int col = t*16 + n;
      hs[(size_t)(rt*16 + g*8 + i)*ADH + col] = (bf16)tanhf(acc[t][i] + b1[col]);
    }
}

__global__ void __launch_bounds__(256)
k_smooth(const bf16* __restrict__ hs, const float* __restrict__ w2,
         const float* __restrict__ b2, float* __restrict__ smooth) {
  int row  = blockIdx.x * (blockDim.x >> 5) + (threadIdx.x >> 5);
  if (row >= ROWS) return;
  int lane = threadIdx.x & 31;
  const bf16* h = hs + (size_t)row * ADH;
  float s = 0.0f;
  #pragma unroll
  for (int i = 0; i < 4; ++i) s += (float)h[lane + 32*i] * w2[lane + 32*i];
  s = wave_sum(s);
  float old = smooth[row];
  if (lane == 0) smooth[row] = fmaxf(0.0f, s + b2[0] + old);
}

// ---------------- launcher ----------------
extern "C" void kernel_launch(void* const* d_in, const int* in_sizes, int n_in,
                              void* d_out, int out_size, void* d_ws, size_t ws_size,
                              hipStream_t stream) {
  const float* query  = (const float*)d_in[0];
  const float* ctx    = (const float*)d_in[1];
  const float* cw_W   = (const float*)d_in[2];
  const float* cw_b   = (const float*)d_in[3];
  const float* sw_W1  = (const float*)d_in[4];
  const float* sw_b1  = (const float*)d_in[5];
  const float* sw_W2  = (const float*)d_in[6];
  const float* sw_b2  = (const float*)d_in[7];
  const float* mw_W1  = (const float*)d_in[8];
  const float* mw_b1  = (const float*)d_in[9];
  const float* mw_W2  = (const float*)d_in[10];
  const float* mw_b2  = (const float*)d_in[11];

  float* out      = (float*)d_out;
  float* out_q    = out;
  float* out_wc   = out + (size_t)BB*NQ*DD;
  float* out_attn = out + (size_t)2*BB*NQ*DD;

  char* ws = (char*)d_ws;
  bf16*  ctxb   = (bf16*)(ws + OFF_CTXB);
  bf16*  ctxT   = (bf16*)(ws + OFF_CTXT);
  bf16*  qm     = (bf16*)(ws + OFF_QM);
  float* matrix = (float*)(ws + OFF_MAT);
  float* T      = (float*)(ws + OFF_T);
  float* rn     = (float*)(ws + OFF_RN);
  float* smooth = (float*)(ws + OFF_SM);
  bf16*  attnb  = (bf16*)(ws + OFF_ATT);
  bf16*  d2     = (bf16*)(ws + OFF_D2);
  float* cp     = (float*)(ws + OFF_CP);
  bf16*  cb     = (bf16*)(ws + OFF_CB);
  bf16*  h1     = (bf16*)(ws + OFF_H1);
  bf16*  hs     = (bf16*)(ws + OFF_HS);
  bf16*  cwT    = (bf16*)(ws + OFF_CWT);
  bf16*  mw1T   = (bf16*)(ws + OFF_MW1T);
  bf16*  mw2T   = (bf16*)(ws + OFF_MW2T);
  bf16*  sw1T   = (bf16*)(ws + OFF_SW1T);

  // state init + operand conversion (every call: no cross-call state)
  k_init<<<1024, 256, 0, stream>>>(matrix, smooth);
  {
    size_t n = (size_t)BB*NC*DD;
    k_cvt_ctx<<<(unsigned)((n + 255)/256), 256, 0, stream>>>(ctx, ctxb, ctxT);
  }
  {
    size_t n = (size_t)NL*(DD*AD + AD*AD2 + AD2*DD + AD*ADH);
    k_cvt_w<<<(unsigned)((n + 255)/256), 256, 0, stream>>>(
        cw_W, mw_W1, mw_W2, sw_W1, cwT, mw1T, mw2T, sw1T);
  }

  const unsigned QMB = (unsigned)(((size_t)BB*NQ*DD + 255)/256);

  for (int layer = 0; layer <= NL; ++layer) {
    bool fin = (layer == NL);
    // attention scan
    k_qm<<<QMB, 256, 0, stream>>>(query, matrix, qm);
    k_scores <<<256,  256, 0, stream>>>(qm, ctxb, T);
    k_rnorm  <<<128,  128, 0, stream>>>(T, rn);
    k_softmax<<<4096, 256, 0, stream>>>(T, rn, smooth, attnb,
                                        fin ? out_attn : (float*)nullptr);
    k_wc     <<<2048, 256, 0, stream>>>(attnb, ctxT, query, d2, out_wc,
                                        fin ? 1 : 0);
    if (fin) break;

    // vectorized similarity + MLPs for this layer
    k_common  <<<512,  256, 0, stream>>>(d2, cwT + (size_t)layer*AD*DD,
                                         cw_b + (size_t)layer*AD, cp);
    k_l2common<<<4096, 256, 0, stream>>>(cp, cb);
    k_h1      <<<1024, 256, 0, stream>>>(cb, mw1T + (size_t)layer*AD2*AD,
                                         mw_b1 + (size_t)layer*AD2, h1);
    k_matrix  <<<2048, 256, 0, stream>>>(h1, mw2T + (size_t)layer*DD*AD2,
                                         mw_b2 + (size_t)layer*DD, matrix);
    k_hs      <<<256,  256, 0, stream>>>(cb, sw1T + (size_t)layer*ADH*AD,
                                         sw_b1 + (size_t)layer*ADH, hs);
    k_smooth  <<<4096, 256, 0, stream>>>(hs, sw_W2 + (size_t)layer*ADH,
                                         sw_b2 + layer, smooth);
  }

  // output 1: query passthrough
  hipMemcpyAsync(out_q, query, (size_t)BB*NQ*DD*sizeof(float),
                 hipMemcpyDeviceToDevice, stream);
}
